// VectorQuantizer_15994458210386
// MI455X (gfx1250) — matrix operand order repaired and checked by reading, not compiled, not run on hardware
//
#include <hip/hip_runtime.h>

typedef __attribute__((ext_vector_type(16))) _Float16 v16h;
typedef __attribute__((ext_vector_type(8)))  float    v8f;

#define NUM_EMB   512
#define EMB_DIM   64
#define HW        4096                 // 64*64
#define NROWS     (32 * HW)            // 131072 spatial vectors
#define Q_ELEMS   (NROWS * EMB_DIM)    // 8388608
#define OUT_VQ    Q_ELEMS
#define OUT_COMMIT (Q_ELEMS + 1)
#define OUT_IDX   (Q_ELEMS + 2)
#define NBLOCKS   (NROWS / 64)         // 2048 blocks, 4 waves each, 16 rows/wave

// d_ws layout (in floats):
//   [WS_NORM  .. +512)  : ||e_k||^2
//   [WS_BFRAG .. +16384): emb as f16 in WMMA B-fragment order (32768 halfs = 64KB)
//   [WS_PART  .. +2048) : per-block partial sums of (q-z)^2
#define WS_NORM   8
#define WS_BFRAG  640     // 2560 bytes offset -> 32B aligned for v16h loads
#define WS_PART   (WS_BFRAG + 16384)

// ---------------------------------------------------------------------------
// Setup: embedding norms + pre-swizzled f16 B fragments.
// B layout for v_wmma_f32_16x16x32_f16 (32x16, K x N):
//   lane l holds column e = tile*16 + (l&15); K = 16*(l>>4) + j  (j = 0..15 f16s)
// Stored fragment-major: frag(t, kstep) -> 32 lanes x 16 halfs contiguous.
// ---------------------------------------------------------------------------
__global__ void vq_setup(const float* __restrict__ emb, float* __restrict__ ws) {
    int i = blockIdx.x * blockDim.x + threadIdx.x;
    if (i < NUM_EMB) {
        float s = 0.f;
        #pragma unroll 8
        for (int d = 0; d < EMB_DIM; ++d) { float v = emb[i * EMB_DIM + d]; s += v * v; }
        ws[WS_NORM + i] = s;
    }
    _Float16* bf = (_Float16*)(ws + WS_BFRAG);
    for (int k = i; k < NUM_EMB * EMB_DIM; k += gridDim.x * blockDim.x) {
        int j = k & 15;
        int l = (k >> 4) & 31;
        int s = (k >> 9) & 1;      // k-step (d 0..31 vs 32..63)
        int t = k >> 10;           // column tile (16 embeddings)
        int e = t * 16 + (l & 15);
        int d = s * 32 + 16 * (l >> 4) + j;
        bf[k] = (_Float16)emb[e * EMB_DIM + d];
    }
}

// ---------------------------------------------------------------------------
// Main: per wave, 16 rows x 512 embeddings via WMMA, fused argmin + gather +
// output write + loss partial. Deterministic (no float atomics).
// B fragments + norms are software-pipelined one tile ahead so the
// s_wait_loadcnt before each WMMA covers a full iteration of latency.
// ---------------------------------------------------------------------------
__global__ void __launch_bounds__(128) vq_main(const float* __restrict__ z,
                                               const float* __restrict__ emb,
                                               const float* __restrict__ ws,
                                               float* __restrict__ out,
                                               float* __restrict__ partials) {
    __shared__ int   lidx[4][16];
    __shared__ float wsum[4];

    const int lane = threadIdx.x & 31;
    const int wv   = threadIdx.x >> 5;
    const int gw   = blockIdx.x * 4 + wv;
    const int n0   = gw * 16;                 // first of this wave's 16 rows
    const int b    = n0 >> 12;                // batch index (4096 rows / batch)
    const int hw0  = n0 & 4095;               // plane offset (16-aligned)
    const int r    = lane & 15;               // row within tile
    const int lh   = lane >> 4;               // lane half

    const float* zb = z + (size_t)b * EMB_DIM * HW;

    // A fragments (16-bit A 16x32, contiguous k, not the ISA table order): lane holds row r,
    // K = j + 16*lh  -> 16 consecutive d per half (matches B packing).
    v16h a0, a1;
    #pragma unroll
    for (int j = 0; j < 16; ++j) {
        int dd = j + 16 * lh;
        a0[j] = (_Float16)zb[(size_t)dd        * HW + hw0 + r];
        a1[j] = (_Float16)zb[(size_t)(dd + 32) * HW + hw0 + r];
    }

    const v16h*  bptr  = (const v16h*)(ws + WS_BFRAG) + lane;
    const float* enorm = ws + WS_NORM;

    float best[8];
    int   bidx[8];
    #pragma unroll
    for (int i = 0; i < 8; ++i) { best[i] = 3.4e38f; bidx[i] = 0; }

    // prologue of the software pipeline: tile 0's operands
    v16h  b0 = bptr[0];
    v16h  b1 = bptr[32];
    float en = enorm[r];

    #pragma unroll 2
    for (int t = 0; t < 32; ++t) {
        // issue next tile's loads now; wrap keeps the address in-bounds and
        // the final (unused) prefetch harmless.
        const int tn = (t + 1) & 31;
        v16h  nb0 = bptr[(tn * 2 + 0) * 32];
        v16h  nb1 = bptr[(tn * 2 + 1) * 32];
        float nen = enorm[tn * 16 + r];

        v8f acc = {};
        acc = __builtin_amdgcn_wmma_f32_16x16x32_f16(false, a0, false, b0,
                                                     (short)0, acc, false, false);
        acc = __builtin_amdgcn_wmma_f32_16x16x32_f16(false, a1, false, b1,
                                                     (short)0, acc, false, false);

        // C/D layout: lane&15 = column (embedding within tile), VGPR i = row i+8*lh
        int e = t * 16 + r;
        #pragma unroll
        for (int i = 0; i < 8; ++i) {
            float sc = en - 2.0f * acc[i];   // ||e||^2 - 2 z.e (||z||^2 constant)
            if (sc < best[i]) { best[i] = sc; bidx[i] = e; }
        }

        b0 = nb0; b1 = nb1; en = nen;
    }

    // argmin across the 16 lanes of each half (xor masks 1..8 stay in-half)
    #pragma unroll
    for (int m = 8; m >= 1; m >>= 1) {
        #pragma unroll
        for (int i = 0; i < 8; ++i) {
            float ob = __shfl_xor(best[i], m, 32);
            int   oi = __shfl_xor(bidx[i], m, 32);
            if (ob < best[i] || (ob == best[i] && oi < bidx[i])) {
                best[i] = ob; bidx[i] = oi;
            }
        }
    }
    // lane 0 holds rows 0..7 (i=0..7), lane 16 holds rows 8..15
    if (r == 0) {
        #pragma unroll
        for (int i = 0; i < 8; ++i) lidx[wv][i + 8 * lh] = bidx[i];
    }
    __syncthreads();

    const int rowidx = lidx[wv][r];
    if (lane < 16) out[OUT_IDX + n0 + lane] = (float)rowidx;   // idx as float

    // Gather exact f32 codebook rows, write quantized in [c][hw] coalesced
    // order (lanes 0-15: c=0..31, lanes 16-31: c=32..63), accumulate loss.
    float* qb = out + (size_t)b * EMB_DIM * HW;
    float  local = 0.f;
    #pragma unroll 8
    for (int cc = 0; cc < 32; ++cc) {
        int    c   = cc + 32 * lh;
        float  qv  = emb[rowidx * EMB_DIM + c];
        size_t off = (size_t)c * HW + hw0 + r;
        float  zv  = zb[off];
        qb[off] = qv;
        float  d = qv - zv;
        local += d * d;
    }
    #pragma unroll
    for (int m = 16; m >= 1; m >>= 1) local += __shfl_xor(local, m, 32);

    if (lane == 0) wsum[wv] = local;
    __syncthreads();
    if (threadIdx.x == 0)
        partials[blockIdx.x] = wsum[0] + wsum[1] + wsum[2] + wsum[3];
}

// ---------------------------------------------------------------------------
// Finalize: deterministic partial-sum reduction, write both loss scalars
// (vq_loss == commitment_loss numerically: both mean((q - z)^2)).
// ---------------------------------------------------------------------------
__global__ void vq_final(const float* __restrict__ ws, float* __restrict__ out) {
    double s = 0.0;
    for (int i = 0; i < NBLOCKS; ++i) s += (double)ws[WS_PART + i];
    float mean = (float)(s / (double)Q_ELEMS);
    out[OUT_VQ]     = mean;
    out[OUT_COMMIT] = mean;
}

extern "C" void kernel_launch(void* const* d_in, const int* in_sizes, int n_in,
                              void* d_out, int out_size, void* d_ws, size_t ws_size,
                              hipStream_t stream) {
    const float* z   = (const float*)d_in[0];   // [32, 64, 64, 64] f32
    const float* emb = (const float*)d_in[1];   // [512, 64] f32
    float* out = (float*)d_out;
    float* ws  = (float*)d_ws;                  // needs ~76 KB of scratch

    vq_setup<<<64, 256, 0, stream>>>(emb, ws);
    vq_main<<<NBLOCKS, 128, 0, stream>>>(z, emb, ws, out, ws);
    vq_final<<<1, 1, 0, stream>>>(ws, out);
}
